// Cond_NCP_30262339568391
// MI455X (gfx1250) — matrix operand order repaired and checked by reading, not compiled, not run on hardware
//
#include <hip/hip_runtime.h>
#include <hip/hip_bf16.h>

typedef __attribute__((ext_vector_type(16))) _Float16 v16h;
typedef __attribute__((ext_vector_type(8)))  float    v8f;

namespace {
constexpr int kT = 512;
constexpr int kB = 128;
constexpr int kOutC = 3;
constexpr int kCtrlElems = kB * kT * kOutC;   // 196608

// workspace byte offsets (d_ws assumed >= 43 MB)
constexpr size_t offW0x = 0;                                  // [160][544] f16
constexpr size_t offW0h = offW0x + size_t(160) * 544 * 2;     // [160][64]  f16
constexpr size_t offW1  = offW0h + size_t(160) * 64 * 2;      // [96][64]   f16
constexpr size_t offW2  = offW1  + size_t(96)  * 64 * 2;      // [16][32]   f16
constexpr size_t offB0  = offW2  + size_t(16)  * 32 * 2;      // [160] f32
constexpr size_t offB1  = offB0  + 160 * 4;                   // [96]  f32
constexpr size_t offB2  = offB1  + 96 * 4;                    // [16]  f32
constexpr size_t offPre = 262144;                             // [65536][160] f32
}

__device__ __forceinline__ float sigm(float x) { return 1.f / (1.f + __expf(-x)); }

// Async global->LDS B128 copy (gfx1250 GLOBAL_LOAD_ASYNC_TO_LDS_B128,
// tracked by ASYNCcnt). Inline asm: portable across ROCm/upstream toolchains.
__device__ __forceinline__ void async_ld16(void* lds, const void* gaddr) {
  unsigned lds_off = (unsigned)(unsigned long long)lds;
  asm volatile("global_load_async_to_lds_b128 %0, %1, off"
               :: "v"(lds_off), "v"(gaddr) : "memory");
}

// ---------------------------------------------------------------------------
// Prep: apply sparsity masks, pack 4 gates per layer row-wise, convert to f16,
// zero-pad K to multiples of 32 so WMMA B-fragments need no bounds checks.
// ---------------------------------------------------------------------------
struct PrepArgs {
  const float* W0[4]; const float* b0[4];
  const float* W1[4]; const float* b1[4];
  const float* W2[4]; const float* b2[4];
  const unsigned char* m0; const unsigned char* m1; const unsigned char* m2;
  _Float16* W0x; _Float16* W0h; _Float16* W1p; _Float16* W2p;
  float* bc0; float* bc1; float* bc2;
};

__global__ void prep_kernel(PrepArgs p) {
  const int tid0   = blockIdx.x * blockDim.x + threadIdx.x;
  const int stride = gridDim.x * blockDim.x;

  for (int idx = tid0; idx < 160 * 544; idx += stride) {
    int w = idx / 544, k = idx - w * 544;
    float v = 0.f;
    if (w < 148 && k < 518) {
      int g = w / 37, rr = w - g * 37;
      v = p.W0[g][rr * 555 + k];
      if (g < 2 && p.m0[rr * 555 + k] == 0) v = 0.f;
    }
    p.W0x[idx] = (_Float16)v;
  }
  for (int idx = tid0; idx < 160 * 64; idx += stride) {
    int w = idx / 64, k = idx - w * 64;
    float v = 0.f;
    if (w < 148 && k < 37) {
      int g = w / 37, rr = w - g * 37;
      int col = 518 + k;
      v = p.W0[g][rr * 555 + col];
      if (g < 2 && p.m0[rr * 555 + col] == 0) v = 0.f;
    }
    p.W0h[idx] = (_Float16)v;
  }
  for (int idx = tid0; idx < 96 * 64; idx += stride) {
    int w = idx / 64, k = idx - w * 64;
    float v = 0.f;
    if (k < 61) {
      int g = w / 24, rr = w - g * 24;
      v = p.W1[g][rr * 61 + k];
      if (g < 2 && p.m1[rr * 61 + k] == 0) v = 0.f;
    }
    p.W1p[idx] = (_Float16)v;
  }
  for (int idx = tid0; idx < 16 * 32; idx += stride) {
    int w = idx / 32, k = idx - w * 32;
    float v = 0.f;
    if (w < 12 && k < 27) {
      int g = w / 3, rr = w - g * 3;
      v = p.W2[g][rr * 27 + k];
      if (g < 2 && p.m2[rr * 27 + k] == 0) v = 0.f;
    }
    p.W2p[idx] = (_Float16)v;
  }
  for (int idx = tid0; idx < 160; idx += stride)
    p.bc0[idx] = (idx < 148) ? p.b0[idx / 37][idx % 37] : 0.f;
  for (int idx = tid0; idx < 96; idx += stride)
    p.bc1[idx] = p.b1[idx / 24][idx % 24];
  for (int idx = tid0; idx < 16; idx += stride)
    p.bc2[idx] = (idx < 12) ? p.b2[idx / 3][idx % 3] : 0.f;
}

// ---------------------------------------------------------------------------
// Time-parallel GEMM: pre0[m][n] = concat(features,command)[m] . W0x[n],
// m = b*512 + t (65536 rows), n = 0..159 (4 gates x 37, padded).
// ---------------------------------------------------------------------------
__global__ __launch_bounds__(256) void gemm0_kernel(
    const float* __restrict__ feat, const float* __restrict__ cmd,
    const _Float16* __restrict__ W0x, float* __restrict__ pre0) {
  const int wave = threadIdx.x >> 5;
  const int lane = threadIdx.x & 31;
  const int r    = lane & 15;
  const int hh   = lane >> 4;
  const int m0   = (blockIdx.x * 8 + wave) * 16;

  const float* frow = feat + (size_t)(m0 + r) * 512;
  const float* crow = cmd  + (size_t)(m0 + r) * 6;

  v8f acc[10] = {};

  for (int kc = 0; kc < 16; ++kc) {
    const int kb = kc * 32;
    v16h a;
#pragma unroll
    for (int j = 0; j < 8; ++j) {
      int k = kb + hh * 16 + (j << 1);
      float2 f = *(const float2*)(frow + k);
      a[2 * j]     = (_Float16)f.x;
      a[2 * j + 1] = (_Float16)f.y;
    }
#pragma unroll
    for (int nt = 0; nt < 10; ++nt) {
      v16h bf = *(const v16h*)(W0x + (size_t)(nt * 16 + r) * 544 + kb + hh * 16);
      acc[nt] = __builtin_amdgcn_wmma_f32_16x16x32_f16(false, a, false, bf,
                                                       (short)0, acc[nt], false, false);
    }
  }
  {
    const int kb = 512;
    v16h a;
#pragma unroll
    for (int i = 0; i < 16; ++i) {
      int k = kb + hh * 16 + i;
      float v = 0.f;
      if (k < 518) v = crow[k - 512];
      a[i] = (_Float16)v;
    }
#pragma unroll
    for (int nt = 0; nt < 10; ++nt) {
      v16h bf = *(const v16h*)(W0x + (size_t)(nt * 16 + r) * 544 + kb + hh * 16);
      acc[nt] = __builtin_amdgcn_wmma_f32_16x16x32_f16(false, a, false, bf,
                                                       (short)0, acc[nt], false, false);
    }
  }
#pragma unroll
  for (int nt = 0; nt < 10; ++nt)
#pragma unroll
    for (int j = 0; j < 8; ++j)
      pre0[(size_t)(m0 + j + 8 * hh) * 160 + nt * 16 + r] = acc[nt][j];
}

// ---------------------------------------------------------------------------
// Sequential scan: 8 blocks x 16 batch rows, 10 waves. pre0 tiles are
// double-buffered in LDS via async global->LDS loads issued one step ahead
// (ASYNCcnt), hiding the HBM/L2 latency behind the 6 compute stages.
// ---------------------------------------------------------------------------
__global__ __launch_bounds__(320) void scan_kernel(
    const float* __restrict__ h0, const float* __restrict__ pre0,
    const float* __restrict__ bc0, const _Float16* __restrict__ W0h,
    const float* __restrict__ bc1, const _Float16* __restrict__ W1p,
    const float* __restrict__ bc2, const _Float16* __restrict__ W2p,
    float* __restrict__ out) {
  __shared__ __align__(16) float s_pre[2][16][160];  // double-buffered pre0 tile
  __shared__ _Float16 s_hi[16][64];
  __shared__ _Float16 s_x1[16][64];
  __shared__ _Float16 s_x2[16][32];
  __shared__ float    s_g[16][160];
  __shared__ float    s_hif[16][37];
  __shared__ float    s_hc[16][24];
  __shared__ float    s_hm[16][3];
  __shared__ float    s_b0[160];
  __shared__ float    s_b1[96];
  __shared__ float    s_b2[16];

  const int tid  = threadIdx.x;
  const int wave = tid >> 5;
  const int lane = tid & 31;
  const int r    = lane & 15;
  const int hh   = lane >> 4;
  const int b0   = blockIdx.x * 16;

  // init state + stage biases in LDS
  for (int i = tid; i < 16 * 64; i += blockDim.x) {
    int m = i >> 6, c = i & 63;
    float v = h0[(b0 + m) * 64 + c];
    if (c < 37) { s_hi[m][c] = (_Float16)v; s_hif[m][c] = v; }
    else {
      s_hi[m][c] = (_Float16)0.f;
      if (c < 61) s_hc[m][c - 37] = v; else s_hm[m][c - 61] = v;
    }
  }
  for (int i = tid; i < 160; i += blockDim.x) s_b0[i] = bc0[i];
  for (int i = tid; i < 96;  i += blockDim.x) s_b1[i] = bc1[i];
  for (int i = tid; i < 16;  i += blockDim.x) s_b2[i] = bc2[i];

  // preload pre0 tile for t=0 into buffer 0 (640 x B128 across 320 threads)
  for (int q = tid; q < 640; q += blockDim.x) {
    int m = q / 40, c = q - m * 40;
    const float* src = pre0 + ((size_t)(b0 + m) * kT + 0) * 160 + c * 4;
    async_ld16(&s_pre[0][m][c * 4], src);
  }
  asm volatile("s_wait_asynccnt 0" ::: "memory");
  __syncthreads();

  for (int t = 0; t < kT; ++t) {
    // -- stage 0: async prefetch pre0 tile for t+1 into the other buffer ----
    {
      const int tt  = (t + 1 < kT) ? (t + 1) : (kT - 1);
      const int buf = (t + 1) & 1;
      for (int q = tid; q < 640; q += blockDim.x) {
        int m = q / 40, c = q - m * 40;
        const float* src = pre0 + ((size_t)(b0 + m) * kT + tt) * 160 + c * 4;
        async_ld16(&s_pre[buf][m][c * 4], src);
      }
      // 2 async ops just issued per wave stay outstanding; waiting <=2
      // guarantees the previous step's copies (buffer t&1) have landed.
      asm volatile("s_wait_asynccnt 2" ::: "memory");
    }
    __syncthreads();

    // -- stage 1: layer0 gates = WMMA(h_inter, W0h) + pre(LDS) + bias -------
    if (wave < 10) {
      v8f acc = {};
#pragma unroll
      for (int kc = 0; kc < 2; ++kc) {
        v16h a;
#pragma unroll
        for (int i = 0; i < 16; ++i)
          a[i] = s_hi[r][kc * 32 + hh * 16 + i];
        v16h bf = *(const v16h*)(W0h + (wave * 16 + r) * 64 + kc * 32 + hh * 16);
        acc = __builtin_amdgcn_wmma_f32_16x16x32_f16(false, a, false, bf,
                                                     (short)0, acc, false, false);
      }
#pragma unroll
      for (int j = 0; j < 8; ++j) {
        int M = j + 8 * hh;
        int N = wave * 16 + r;
        s_g[M][N] = acc[j] + s_pre[t & 1][M][N] + s_b0[N];
      }
    }
    __syncthreads();
    // -- stage 2: CfC activation for layer0, build layer1 input -------------
    for (int i = tid; i < 16 * 64; i += blockDim.x) {
      int m = i >> 6, c = i & 63;
      _Float16 val = (_Float16)0.f;
      if (c < 37) {
        float s  = sigm(s_g[m][74 + c] + s_g[m][111 + c]);
        float hn = tanhf(s_g[m][c]) * (1.f - s) + s * tanhf(s_g[m][37 + c]);
        s_hif[m][c] = hn;
        val = (_Float16)hn;
      } else if (c < 61) {
        val = (_Float16)s_hc[m][c - 37];
      }
      s_x1[m][c] = val;
    }
    __syncthreads();
    // -- stage 3: layer1 gates via WMMA -------------------------------------
    if (wave < 6) {
      v8f acc = {};
#pragma unroll
      for (int kc = 0; kc < 2; ++kc) {
        v16h a;
#pragma unroll
        for (int i = 0; i < 16; ++i)
          a[i] = s_x1[r][kc * 32 + hh * 16 + i];
        v16h bf = *(const v16h*)(W1p + (wave * 16 + r) * 64 + kc * 32 + hh * 16);
        acc = __builtin_amdgcn_wmma_f32_16x16x32_f16(false, a, false, bf,
                                                     (short)0, acc, false, false);
      }
#pragma unroll
      for (int j = 0; j < 8; ++j)
        s_g[j + 8 * hh][wave * 16 + r] = acc[j] + s_b1[wave * 16 + r];
    }
    __syncthreads();
    // -- stage 4: CfC activation for layer1, build layer2 input -------------
    for (int i = tid; i < 16 * 32; i += blockDim.x) {
      int m = i >> 5, c = i & 31;
      _Float16 val = (_Float16)0.f;
      if (c < 24) {
        float s  = sigm(s_g[m][48 + c] + s_g[m][72 + c]);
        float hn = tanhf(s_g[m][c]) * (1.f - s) + s * tanhf(s_g[m][24 + c]);
        s_hc[m][c] = hn;
        val = (_Float16)hn;
      } else if (c < 27) {
        val = (_Float16)s_hm[m][c - 24];
      }
      s_x2[m][c] = val;
    }
    __syncthreads();
    // -- stage 5: layer2 gates via WMMA (single wave, K=32, N=12) -----------
    if (wave == 0) {
      v8f acc = {};
      v16h a;
#pragma unroll
      for (int i = 0; i < 16; ++i)
        a[i] = s_x2[r][hh * 16 + i];
      v16h bf = *(const v16h*)(W2p + r * 32 + hh * 16);
      acc = __builtin_amdgcn_wmma_f32_16x16x32_f16(false, a, false, bf,
                                                   (short)0, acc, false, false);
#pragma unroll
      for (int j = 0; j < 8; ++j)
        s_g[j + 8 * hh][r] = acc[j] + s_b2[r];
    }
    __syncthreads();
    // -- stage 6: layer2 activation, emit controls, roll state --------------
    for (int i = tid; i < 16 * 64; i += blockDim.x) {
      int m = i >> 6, c = i & 63;
      if (c < 3) {
        float s = sigm(s_g[m][6 + c] + s_g[m][9 + c]);
        float o = tanhf(s_g[m][c]) * (1.f - s) + s * tanhf(s_g[m][3 + c]);
        s_hm[m][c] = o;
        float ctrl = (c == 0) ? tanhf(o) : sigm(o);
        out[((size_t)(b0 + m) * kT + t) * kOutC + c] = ctrl;
      }
      s_hi[m][c] = (c < 37) ? s_x1[m][c] : (_Float16)0.f;
    }
    __syncthreads();
  }

  // h_final = [h_inter | h_cmd | h_motor]
  for (int i = tid; i < 16 * 64; i += blockDim.x) {
    int m = i >> 6, c = i & 63;
    float v = (c < 37) ? s_hif[m][c] : (c < 61) ? s_hc[m][c - 37] : s_hm[m][c - 61];
    out[kCtrlElems + (b0 + m) * 64 + c] = v;
  }
}

// ---------------------------------------------------------------------------
extern "C" void kernel_launch(void* const* d_in, const int* in_sizes, int n_in,
                              void* d_out, int out_size, void* d_ws, size_t ws_size,
                              hipStream_t stream) {
  (void)in_sizes; (void)n_in; (void)out_size; (void)ws_size;

  const float* feat = (const float*)d_in[0];
  const float* cmd  = (const float*)d_in[1];
  const float* h0   = (const float*)d_in[2];

  char* ws = (char*)d_ws;
  _Float16* W0x = (_Float16*)(ws + offW0x);
  _Float16* W0h = (_Float16*)(ws + offW0h);
  _Float16* W1p = (_Float16*)(ws + offW1);
  _Float16* W2p = (_Float16*)(ws + offW2);
  float* bc0 = (float*)(ws + offB0);
  float* bc1 = (float*)(ws + offB1);
  float* bc2 = (float*)(ws + offB2);
  float* pre0 = (float*)(ws + offPre);

  PrepArgs pa;
  for (int g = 0; g < 4; ++g) {
    pa.W0[g] = (const float*)d_in[6 + g * 2];
    pa.b0[g] = (const float*)d_in[7 + g * 2];
    pa.W1[g] = (const float*)d_in[14 + g * 2];
    pa.b1[g] = (const float*)d_in[15 + g * 2];
    pa.W2[g] = (const float*)d_in[22 + g * 2];
    pa.b2[g] = (const float*)d_in[23 + g * 2];
  }
  pa.m0 = (const unsigned char*)d_in[3];
  pa.m1 = (const unsigned char*)d_in[4];
  pa.m2 = (const unsigned char*)d_in[5];
  pa.W0x = W0x; pa.W0h = W0h; pa.W1p = W1p; pa.W2p = W2p;
  pa.bc0 = bc0; pa.bc1 = bc1; pa.bc2 = bc2;

  prep_kernel<<<120, 256, 0, stream>>>(pa);
  gemm0_kernel<<<512, 256, 0, stream>>>(feat, cmd, W0x, pre0);
  scan_kernel<<<8, 320, 0, stream>>>(h0, pre0, bc0, W0h, bc1, W1p, bc2, W2p,
                                     (float*)d_out);
}
